// DTODRLBackbone_64544768524833
// MI455X (gfx1250) — matrix operand order repaired and checked by reading, compiled, not run on hardware
//
#include <hip/hip_runtime.h>
#include <hip/hip_bf16.h>

typedef __attribute__((ext_vector_type(16))) _Float16 v16h;
typedef __attribute__((ext_vector_type(8)))  _Float16 v8h;
typedef __attribute__((ext_vector_type(8)))  float    v8f;

#define NNODES 50000
#define MPAD   50048      // multiple of 128 (8 waves x 16 rows)
#define ERAW   800000
#define ETOT   850000     // + self loops
#define NGRAPH 64
#define NLOCS  50
#define KPADMAX 160
#define NTMAX   9
#define KTMAX   5

// ---------------- device helpers ----------------
__device__ __forceinline__ float leaky02(float v) { return v > 0.f ? v : 0.2f * v; }

// order-preserving float -> uint encoding for atomicMax
__device__ __forceinline__ unsigned fenc(float f) {
    unsigned u = __float_as_uint(f);
    return (u & 0x80000000u) ? ~u : (u | 0x80000000u);
}
__device__ __forceinline__ float fdec(unsigned e) {
    unsigned u = (e & 0x80000000u) ? (e & 0x7FFFFFFFu) : ~e;
    return __uint_as_float(u);
}
#define ENC_NEGINF 0x007FFFFFu   // fenc(-inf)

__device__ __forceinline__ void edge_ends(int i, const int* __restrict__ src,
                                          const int* __restrict__ dst, int& s, int& d) {
    if (i < ERAW) { s = src[i]; d = dst[i]; }
    else          { s = i - ERAW; d = s; }          // self loops
}

// ---------------- fill kernels ----------------
__global__ void k_fill_f32(float* p, float v, int n) {
    int i = blockIdx.x * blockDim.x + threadIdx.x;
    if (i < n) p[i] = v;
}
__global__ void k_fill_u32(unsigned* p, unsigned v, int n) {
    int i = blockIdx.x * blockDim.x + threadIdx.x;
    if (i < n) p[i] = v;
}

// ---------------- cast activations f32 -> padded f16 ----------------
__global__ void k_cast_h16(const float* __restrict__ h, _Float16* __restrict__ h16,
                           int Kin, int kpad) {
    int idx = blockIdx.x * blockDim.x + threadIdx.x;
    int tot = MPAD * kpad;
    if (idx >= tot) return;
    int n = idx / kpad, k = idx - n * kpad;
    float v = (n < NNODES && k < Kin) ? h[(size_t)n * Kin + k] : 0.f;
    h16[idx] = (_Float16)v;
}

// ---------------- pack weights into per-lane WMMA B-fragment layout ----------------
// layout: idx = ((nTile*KT + kTile)*32 + lane)*16 + e ; K = kTile*32 + (e>>3)*16 + laneHi*8 + (e&7) ; N = nTile*16 + laneLow
__global__ void k_pack_w(const float* __restrict__ W, _Float16* __restrict__ Bp,
                         int Kin, int Nout, int KT, int NT) {
    int idx = blockIdx.x * blockDim.x + threadIdx.x;
    int tot = KT * NT * 512;
    if (idx >= tot) return;
    int e    = idx & 15;
    int lane = (idx >> 4) & 31;
    int t    = idx >> 9;               // nTile*KT + kTile
    int kTile = t % KT, nTile = t / KT;
    int laneLow = lane & 15, laneHi = lane >> 4;
    int k = kTile * 32 + (e >> 3) * 16 + laneHi * 8 + (e & 7);
    int n = nTile * 16 + laneLow;
    float v = (k < Kin && n < Nout) ? W[(size_t)k * Nout + n] : 0.f;
    Bp[idx] = (_Float16)v;
}

// ---------------- WMMA GEMM: C[M,Nout] = A16[M,kpad] x Wpack ----------------
__global__ void k_wmma_gemm(const _Float16* __restrict__ A, const _Float16* __restrict__ Bp,
                            float* __restrict__ C, int kpad, int KT, int Nout, int ldc) {
    int lane = threadIdx.x & 31;
    int wave = threadIdx.x >> 5;
    int nTile = blockIdx.x;
    int mTile = blockIdx.y * 8 + wave;
    int laneLow = lane & 15, laneHi = lane >> 4;

    int row = mTile * 16 + laneLow;                 // < MPAD by grid construction
    const _Float16* arow = A + (size_t)row * kpad + laneHi * 8;
    const v16h* bptr = (const v16h*)(Bp + (size_t)(nTile * KT) * 512) + lane;

    v8f acc = {};
    for (int kt = 0; kt < KT; ++kt) {
        const _Float16* pa = arow + kt * 32;
        v8h lo = *(const v8h*)(pa);                 // K = base + 0..7
        v8h hi = *(const v8h*)(pa + 16);            // K = base + 16..23
        v16h a;
#pragma unroll
        for (int i = 0; i < 8; ++i) { a[i] = lo[i]; a[i + 8] = hi[i]; }
        v16h b = *(const v16h*)(bptr + (size_t)kt * 32);
        acc = __builtin_amdgcn_wmma_f32_16x16x32_f16(false, a, false, b,
                                                     (short)0, acc, false, false);
    }
    int ncol = nTile * 16 + laneLow;
#pragma unroll
    for (int r = 0; r < 8; ++r) {
        int mrow = mTile * 16 + laneHi * 8 + r;
        if (mrow < NNODES && ncol < Nout)
            C[(size_t)mrow * ldc + ncol] = acc[r];
    }
}

// ---------------- per-node attention coefficients ----------------
__global__ void k_attn_coeff(const float* __restrict__ Wh, const float* __restrict__ a_src,
                             const float* __restrict__ a_dst, float* __restrict__ es,
                             float* __restrict__ ed, int H, int O, int F) {
    int idx = blockIdx.x * blockDim.x + threadIdx.x;
    if (idx >= NNODES * H) return;
    int n = idx / H, h = idx - n * H;
    const float* hr  = Wh    + (size_t)n * F + h * O;
    const float* asr = a_src + (size_t)h * O;
    const float* adr = a_dst + (size_t)h * O;
    float s = 0.f, d = 0.f;
    for (int o = 0; o < O; ++o) { float v = hr[o]; s += v * asr[o]; d += v * adr[o]; }
    es[idx] = s; ed[idx] = d;
}

// ---------------- segment max over dst ----------------
__global__ void k_edge_max(const int* __restrict__ src, const int* __restrict__ dst,
                           const float* __restrict__ es, const float* __restrict__ ed,
                           unsigned* __restrict__ menc, int H) {
    int idx = blockIdx.x * blockDim.x + threadIdx.x;
    if (idx >= ETOT * H) return;
    int i = idx / H, h = idx - i * H;
    int s, d; edge_ends(i, src, dst, s, d);
    float e = leaky02(es[s * H + h] + ed[d * H + h]);
    atomicMax(&menc[d * H + h], fenc(e));
}

// ---------------- segment sum of exp(e - m) ----------------
__global__ void k_edge_expsum(const int* __restrict__ src, const int* __restrict__ dst,
                              const float* __restrict__ es, const float* __restrict__ ed,
                              const unsigned* __restrict__ menc, float* __restrict__ z, int H) {
    int idx = blockIdx.x * blockDim.x + threadIdx.x;
    if (idx >= ETOT * H) return;
    int i = idx / H, h = idx - i * H;
    int s, d; edge_ends(i, src, dst, s, d);
    float e = leaky02(es[s * H + h] + ed[d * H + h]);
    float m = fdec(menc[d * H + h]);
    if (!isfinite(m)) m = 0.f;
    atomicAdd(&z[d * H + h], __expf(e - m));
}

// ---------------- alpha-weighted scatter-add (one block per edge) ----------------
__global__ void k_edge_scatter(const int* __restrict__ src, const int* __restrict__ dst,
                               const float* __restrict__ es, const float* __restrict__ ed,
                               const unsigned* __restrict__ menc, const float* __restrict__ z,
                               const float* __restrict__ Wh, float* __restrict__ agg,
                               int H, int O, int F) {
    __shared__ float alpha[4];
    int i = blockIdx.x;
    int s, d; edge_ends(i, src, dst, s, d);
    if ((int)threadIdx.x < H) {
        int h = threadIdx.x;
        float e = leaky02(es[s * H + h] + ed[d * H + h]);
        float m = fdec(menc[d * H + h]);
        if (!isfinite(m)) m = 0.f;
        float w = __expf(e - m);
        alpha[h] = w / (z[d * H + h] + 1e-16f);
    }
    __syncthreads();
    int f = threadIdx.x;
    if (f < F) {
        int h = f / O;
        atomicAdd(&agg[(size_t)d * F + f], Wh[(size_t)s * F + f] * alpha[h]);
    }
}

// ---------------- bias + optional leaky (in place) ----------------
__global__ void k_finalize(float* __restrict__ agg, const float* __restrict__ b,
                           int F, int do_leaky) {
    int idx = blockIdx.x * blockDim.x + threadIdx.x;
    if (idx >= NNODES * F) return;
    int f = idx % F;
    float v = agg[idx] + b[f];
    agg[idx] = do_leaky ? leaky02(v) : v;
}

// ---------------- global mean pool over graphs ----------------
__global__ void k_pool_acc(const float* __restrict__ h, const int* __restrict__ batch,
                           float* __restrict__ psum, float* __restrict__ pcnt) {
    int idx = blockIdx.x * blockDim.x + threadIdx.x;
    if (idx >= NNODES * 128) return;
    int n = idx >> 7, f = idx & 127;
    int g = batch[n];
    atomicAdd(&psum[g * 128 + f], h[(size_t)n * 128 + f]);
    if (f == 0) atomicAdd(&pcnt[g], 1.0f);
}
__global__ void k_pool_write(const float* __restrict__ psum, const float* __restrict__ pcnt,
                             float* __restrict__ out) {
    int idx = blockIdx.x * blockDim.x + threadIdx.x;
    if (idx >= NGRAPH * 128) return;
    int g = idx >> 7, f = idx & 127;
    out[g * 256 + f] = psum[idx] / fmaxf(pcnt[g], 1.0f);
}

// ---------------- location encoder MLP (one block per graph) ----------------
__global__ void k_loc_mlp(const float* __restrict__ loc, const float* __restrict__ Wl1,
                          const float* __restrict__ bl1, const float* __restrict__ Wl2,
                          const float* __restrict__ bl2, float* __restrict__ out) {
    __shared__ float hid[256];
    int g = blockIdx.x;
    int t = threadIdx.x;
    float acc = 0.f;
    for (int l = 0; l < NLOCS; ++l) {
        float x0 = loc[((size_t)g * NLOCS + l) * 2 + 0];
        float x1 = loc[((size_t)g * NLOCS + l) * 2 + 1];
        hid[t] = tanhf(x0 * Wl1[t] + x1 * Wl1[256 + t] + bl1[t]);
        __syncthreads();
        if (t < 128) {
            float s = 0.f;
            for (int j = 0; j < 256; ++j) s += hid[j] * Wl2[j * 128 + t];
            acc += s;
        }
        __syncthreads();
    }
    if (t < 128) out[g * 256 + 128 + t] = acc * (1.0f / NLOCS) + bl2[t];
}

// ---------------- host launcher ----------------
extern "C" void kernel_launch(void* const* d_in, const int* in_sizes, int n_in,
                              void* d_out, int out_size, void* d_ws, size_t ws_size,
                              hipStream_t stream) {
    (void)in_sizes; (void)n_in; (void)out_size; (void)ws_size;

    const float* x    = (const float*)d_in[0];
    const float* loc  = (const float*)d_in[1];
    const int*   ei   = (const int*)  d_in[2];
    const int*   batch= (const int*)  d_in[3];
    const float* W0   = (const float*)d_in[4];
    const float* as0  = (const float*)d_in[5];
    const float* ad0  = (const float*)d_in[6];
    const float* b0   = (const float*)d_in[7];
    const float* W1   = (const float*)d_in[8];
    const float* as1  = (const float*)d_in[9];
    const float* ad1  = (const float*)d_in[10];
    const float* b1   = (const float*)d_in[11];
    const float* W2   = (const float*)d_in[12];
    const float* as2  = (const float*)d_in[13];
    const float* ad2  = (const float*)d_in[14];
    const float* b2   = (const float*)d_in[15];
    const float* Wl1  = (const float*)d_in[16];
    const float* bl1  = (const float*)d_in[17];
    const float* Wl2  = (const float*)d_in[18];
    const float* bl2  = (const float*)d_in[19];
    float* out = (float*)d_out;

    const int* e_src = ei;
    const int* e_dst = ei + ERAW;

    // ---- workspace carve-up (all 256B aligned) ----
    size_t off = 0;
    auto take = [&](size_t bytes) -> void* {
        void* p = (char*)d_ws + off;
        off = (off + bytes + 255) & ~(size_t)255;
        return p;
    };
    float*     bufA  = (float*)take((size_t)NNODES * 129 * sizeof(float));
    float*     bufB  = (float*)take((size_t)NNODES * 129 * sizeof(float));
    float*     bufC  = (float*)take((size_t)NNODES * 129 * sizeof(float));
    _Float16*  h16   = (_Float16*)take((size_t)MPAD * KPADMAX * sizeof(_Float16));
    _Float16*  wpack = (_Float16*)take((size_t)KTMAX * NTMAX * 512 * sizeof(_Float16));
    float*     es    = (float*)take((size_t)NNODES * 3 * sizeof(float));
    float*     edv   = (float*)take((size_t)NNODES * 3 * sizeof(float));
    unsigned*  menc  = (unsigned*)take((size_t)NNODES * 3 * sizeof(unsigned));
    float*     z     = (float*)take((size_t)NNODES * 3 * sizeof(float));
    float*     psum  = (float*)take((size_t)NGRAPH * 128 * sizeof(float));
    float*     pcnt  = (float*)take((size_t)NGRAPH * sizeof(float));

    auto run_layer = [&](const float* hin, int Kin, const float* W, const float* a_s,
                         const float* a_d, const float* b, int H, int O,
                         float* Wh, float* agg, int do_leaky) {
        const int F    = H * O;
        const int kpad = (Kin + 31) & ~31;
        const int KT   = kpad / 32;
        const int NT   = (F + 15) / 16;

        int tot = MPAD * kpad;
        k_cast_h16<<<(tot + 255) / 256, 256, 0, stream>>>(hin, h16, Kin, kpad);
        int wtot = KT * NT * 512;
        k_pack_w<<<(wtot + 255) / 256, 256, 0, stream>>>(W, wpack, Kin, F, KT, NT);

        dim3 gg(NT, MPAD / 128);
        k_wmma_gemm<<<gg, 256, 0, stream>>>(h16, wpack, Wh, kpad, KT, F, F);

        k_attn_coeff<<<(NNODES * H + 255) / 256, 256, 0, stream>>>(Wh, a_s, a_d, es, edv, H, O, F);

        k_fill_u32<<<(NNODES * H + 255) / 256, 256, 0, stream>>>(menc, ENC_NEGINF, NNODES * H);
        k_fill_f32<<<(NNODES * H + 255) / 256, 256, 0, stream>>>(z, 0.f, NNODES * H);
        k_fill_f32<<<(NNODES * F + 255) / 256, 256, 0, stream>>>(agg, 0.f, NNODES * F);

        k_edge_max   <<<(ETOT * H + 255) / 256, 256, 0, stream>>>(e_src, e_dst, es, edv, menc, H);
        k_edge_expsum<<<(ETOT * H + 255) / 256, 256, 0, stream>>>(e_src, e_dst, es, edv, menc, z, H);

        int bt = ((F + 31) / 32) * 32;   // 160 for F=129, 128 for F=128
        k_edge_scatter<<<ETOT, bt, 0, stream>>>(e_src, e_dst, es, edv, menc, z, Wh, agg, H, O, F);

        k_finalize<<<(NNODES * F + 255) / 256, 256, 0, stream>>>(agg, b, F, do_leaky);
    };

    // GAT layer 0: x[50000,6] -> bufC[50000,129]  (Wh scratch = bufB)
    run_layer(x,    6,   W0, as0, ad0, b0, 3, 43, bufB, bufC, 1);
    // GAT layer 1: bufC -> bufB                    (Wh scratch = bufA)
    run_layer(bufC, 129, W1, as1, ad1, b1, 3, 43, bufA, bufB, 1);
    // GAT layer 2 (H=1, concat=False == identity mean): bufB -> bufC[50000,128]
    run_layer(bufB, 129, W2, as2, ad2, b2, 1, 128, bufA, bufC, 0);

    // global mean pool -> out[:, 0:128]
    k_fill_f32<<<(NGRAPH * 128 + 255) / 256, 256, 0, stream>>>(psum, 0.f, NGRAPH * 128);
    k_fill_f32<<<1, 64, 0, stream>>>(pcnt, 0.f, NGRAPH);
    k_pool_acc<<<(NNODES * 128 + 255) / 256, 256, 0, stream>>>(bufC, batch, psum, pcnt);
    k_pool_write<<<(NGRAPH * 128 + 255) / 256, 256, 0, stream>>>(psum, pcnt, out);

    // location encoder MLP -> out[:, 128:256]
    k_loc_mlp<<<NGRAPH, 256, 0, stream>>>(loc, Wl1, bl1, Wl2, bl2, out);
}
